// MultiHeadAttention_5677946765667
// MI455X (gfx1250) — matrix operand order repaired and checked by reading, not compiled, not run on hardware
//
#include <hip/hip_runtime.h>

// ---------------------------------------------------------------------------
// Multi-head attention forward for MI455X (gfx1250), wave32 + WMMA bf16 + TDM.
//   B=4, S=2048, D_MODEL=1024, H=16, DK=64
// Pipeline:
//   1) gemm128<f32 in, split-head bf16 out> : Q,K,V projections
//   2) attn_kernel                          : flash-style attention
//                                             (TDM K-stage, bf16 WMMA)
//   3) gemm128<bf16 in, f32 out>            : output projection -> d_out
// ---------------------------------------------------------------------------

typedef __bf16 bf16;
typedef bf16     bf16x16 __attribute__((ext_vector_type(16)));
typedef bf16     bf16x4  __attribute__((ext_vector_type(4)));
typedef float    floatx8 __attribute__((ext_vector_type(8)));
typedef float    floatx4 __attribute__((ext_vector_type(4)));
typedef unsigned uintx4  __attribute__((ext_vector_type(4)));
typedef int      intx4   __attribute__((ext_vector_type(4)));
typedef int      intx8   __attribute__((ext_vector_type(8)));

union FragU { bf16x16 v; unsigned u[8]; };

__device__ __forceinline__ floatx8 zero8() {
  floatx8 z;
#pragma unroll
  for (int i = 0; i < 8; ++i) z[i] = 0.0f;
  return z;
}

// D = A(16x32 bf16) * B(32x16 bf16) + C(16x16 f32)
__device__ __forceinline__ floatx8 wmma_bf16(bf16x16 a, bf16x16 b, floatx8 c) {
  return __builtin_amdgcn_wmma_f32_16x16x32_bf16(false, a, false, b,
                                                 (short)0, c, false, false);
}

// A-fragment (16x32, MxK): lane = 16*grp + m holds row m.
//   VGPR v<4 : K = 2v + 8*grp ; VGPR v>=4: K = 16 + 2(v-4) + 8*grp
// Consecutive k0 per 4 VGPRs -> compiler merges to 2x ds_load_b128.
__device__ __forceinline__ bf16x16 load_frag_a(const bf16* p, int ld) {
  const int lane = threadIdx.x & 31;
  const int grp = lane >> 4, m = lane & 15;
  FragU f;
#pragma unroll
  for (int v = 0; v < 8; ++v) {
    const int k0 = ((v < 4) ? (2 * v) : (16 + 2 * (v - 4))) + grp * 8;
    f.u[v] = *(const unsigned*)(p + (size_t)m * ld + k0);
  }
  return f.v;
}

// B-fragment (32x16, KxN): lane = 16*grp + n holds column n;
//   VGPR v<4 : K = 2v + 8*grp ; VGPR v>=4: K = 16 + 2(v-4) + 8*grp
__device__ __forceinline__ bf16x16 load_frag_b(const bf16* p, int ld) {
  const int lane = threadIdx.x & 31;
  const int grp = lane >> 4, n = lane & 15;
  FragU f;
#pragma unroll
  for (int v = 0; v < 8; ++v) {
    const int k0 = ((v < 4) ? (2 * v) : (16 + 2 * (v - 4))) + grp * 8;
    f.u[v] = *(const unsigned*)(p + (size_t)n * ld + k0);
  }
  return f.v;
}

// ---------------------------------------------------------------------------
// Tensor Data Mover: load a 64x64 bf16 tile (row stride 64 elems, tensor
// height 2048) from global into LDS.  D# built per CDNA5 ISA.
// 6-arg builtin form (clang-23 / therock-10.0 headers).
// ---------------------------------------------------------------------------
__device__ __forceinline__ void tdm_load_tile64(const bf16* gsrc, bf16* ldst) {
  const unsigned long long ga = (unsigned long long)(size_t)gsrc;
  const unsigned la = (unsigned)(size_t)ldst;     // low 32 bits = LDS offset
  uintx4 g0;
  g0[0] = 1u;                                     // count=1, user mode
  g0[1] = la;                                     // lds_addr
  g0[2] = (unsigned)ga;                           // global_addr[31:0]
  g0[3] = (unsigned)((ga >> 32) & 0x1FFFFFFu) | (2u << 30);  // addr[56:32]|type=2
  intx8 g1;
  g1[0] = (int)(1u << 16);        // workgroup_mask=0, data_size=1 (2 bytes)
  g1[1] = (int)(64u << 16);       // tensor_dim0 = 64 (low 16)
  g1[2] = (int)(2048u << 16);     // tensor_dim0 hi=0, tensor_dim1 = 2048 (low 16)
  g1[3] = (int)(64u << 16);       // tensor_dim1 hi=0, tile_dim0 = 64
  g1[4] = (int)64u;               // tile_dim1 = 64, tile_dim2 = 0
  g1[5] = (int)64u;               // tensor_dim0_stride = 64
  g1[6] = 0;                      // stride hi / dim1_stride = 0
  g1[7] = 0;
  intx4 z4;
  z4[0] = 0; z4[1] = 0; z4[2] = 0; z4[3] = 0;
  intx8 z8;
#pragma unroll
  for (int i = 0; i < 8; ++i) z8[i] = 0;
  __builtin_amdgcn_tensor_load_to_lds(g0, g1, z4, z4, z8, 0);
}

// ---------------------------------------------------------------------------
// GEMM: C[M,N] = A[M,K] * W[N,K]^T + bias   (torch nn.Linear convention)
// 256 threads (8 waves), block tile 128x128, K-step 32, wave tile 32x64
// ---------------------------------------------------------------------------
template <bool A_BF16, bool SPLIT_HEADS>
__global__ __launch_bounds__(256) void gemm128(const void* __restrict__ Ap,
                                               const float* __restrict__ W,
                                               const float* __restrict__ bias,
                                               void* __restrict__ Cp,
                                               int M, int N, int K) {
  constexpr int LDK = 40;                 // padded K stride (8B-aligned rows)
  __shared__ bf16 As[128 * LDK];
  __shared__ bf16 Bs[128 * LDK];

  const int tid = threadIdx.x;
  const int wave = tid >> 5;
  const int wm = (wave >> 1) * 32;        // 0,32,64,96
  const int wn = (wave & 1) * 64;         // 0,64
  const int mTile = blockIdx.y * 128;
  const int nTile = blockIdx.x * 128;

  floatx8 acc[2][4];
#pragma unroll
  for (int i = 0; i < 2; ++i)
#pragma unroll
    for (int j = 0; j < 4; ++j) acc[i][j] = zero8();

  for (int k0 = 0; k0 < K; k0 += 32) {
    // ---- stage A tile (128x32) into LDS as bf16 (packed converts) ----
    if constexpr (!A_BF16) {
      const float* A = (const float*)Ap;
#pragma unroll
      for (int it = 0; it < 4; ++it) {
        const int idx = (tid + it * 256) * 4;      // 0..4095
        const int r = idx >> 5, c = idx & 31;
        const floatx4 d = *(const floatx4*)(A + (size_t)(mTile + r) * K + k0 + c);
        *(bf16x4*)&As[r * LDK + c] = __builtin_convertvector(d, bf16x4);
      }
    } else {
      const bf16* A = (const bf16*)Ap;
#pragma unroll
      for (int it = 0; it < 4; ++it) {
        const int idx = (tid + it * 256) * 4;
        const int r = idx >> 5, c = idx & 31;
        *(uint2*)&As[r * LDK + c] =
            *(const uint2*)(A + (size_t)(mTile + r) * K + k0 + c);
      }
    }
    // ---- stage W tile (128 n-rows x 32 k) into LDS as bf16 ----
#pragma unroll
    for (int it = 0; it < 4; ++it) {
      const int idx = (tid + it * 256) * 4;
      const int r = idx >> 5, c = idx & 31;
      const floatx4 d = *(const floatx4*)(W + (size_t)(nTile + r) * K + k0 + c);
      *(bf16x4*)&Bs[r * LDK + c] = __builtin_convertvector(d, bf16x4);
    }
    __syncthreads();

    bf16x16 a[2], b[4];
#pragma unroll
    for (int i = 0; i < 2; ++i) a[i] = load_frag_a(&As[(wm + i * 16) * LDK], LDK);
#pragma unroll
    for (int j = 0; j < 4; ++j) b[j] = load_frag_b(&Bs[(wn + j * 16) * LDK], LDK);
#pragma unroll
    for (int i = 0; i < 2; ++i)
#pragma unroll
      for (int j = 0; j < 4; ++j) acc[i][j] = wmma_bf16(a[i], b[j], acc[i][j]);
    __syncthreads();
  }

  // ---- store ----
  const int lane = tid & 31;
  const int grp = lane >> 4, nlo = lane & 15;
#pragma unroll
  for (int i = 0; i < 2; ++i) {
#pragma unroll
    for (int j = 0; j < 4; ++j) {
      const int gn = nTile + wn + j * 16 + nlo;
      const float bval = bias[gn];
#pragma unroll
      for (int r = 0; r < 8; ++r) {
        const int gm = mTile + wm + i * 16 + r + grp * 8;
        const float v = acc[i][j][r] + bval;
        if constexpr (SPLIT_HEADS) {
          // row gm = b*2048 + s ; col gn = h*64 + dk -> [B,H,S,64] bf16
          const int b = gm >> 11, s = gm & 2047;
          const int h = gn >> 6,  dk = gn & 63;
          ((bf16*)Cp)[(((size_t)(b * 16 + h) * 2048 + s) << 6) + dk] = (bf16)v;
        } else {
          ((float*)Cp)[(size_t)gm * N + gn] = v;
        }
      }
    }
  }
}

// ---------------------------------------------------------------------------
// Attention: per (b,h) head, flash-style streaming over 64-key blocks.
// 128 threads / 4 waves; wave owns 16 query rows; block owns 64 query rows.
// K block staged by the Tensor Data Mover; V staged + transposed manually.
// ---------------------------------------------------------------------------
__global__ __launch_bounds__(128) void attn_kernel(const bf16* __restrict__ Q,
                                                   const bf16* __restrict__ Km,
                                                   const bf16* __restrict__ V,
                                                   bf16* __restrict__ O) {
  constexpr int S = 2048, DK = 64;
  constexpr int PLD = 72;                       // P staging row stride
  __shared__ bf16 Ks[64 * 64];                  // [key][dk]   (TDM target)
  __shared__ bf16 Vt[64 * 64];                  // [dk][key]   (transposed)
  __shared__ bf16 Ps[4][16 * PLD];              // per-wave P staging

  const int tid = threadIdx.x;
  const int wave = tid >> 5, lane = tid & 31;
  const int grp = lane >> 4, nlo = lane & 15;
  const int bh = blockIdx.y;                    // b*16 + h
  const int qb = blockIdx.x;                    // query block (64 rows)

  const bf16* Qbase = Q + ((size_t)bh * S + qb * 64 + wave * 16) * DK;
  const bf16* Kbase = Km + (size_t)bh * S * DK;
  const bf16* Vbase = V  + (size_t)bh * S * DK;

  // Q fragments (held for the whole kernel): dk 0..31 and 32..63
  const bf16x16 qa0 = load_frag_a(Qbase, DK);
  const bf16x16 qa1 = load_frag_a(Qbase + 32, DK);

  float rm[8], rl[8];
  floatx8 acc[4];
#pragma unroll
  for (int r = 0; r < 8; ++r) { rm[r] = -1e30f; rl[r] = 0.0f; }
#pragma unroll
  for (int j = 0; j < 4; ++j) acc[j] = zero8();

  for (int kb = 0; kb < S; kb += 64) {
    // ---- K block -> LDS via Tensor Data Mover (wave 0 issues) ----
    if (wave == 0) tdm_load_tile64(Kbase + (size_t)kb * DK, Ks);

    // ---- V block -> LDS transposed [dk][key] (all threads) ----
#pragma unroll
    for (int it = 0; it < 8; ++it) {
      const int idx = (tid + it * 128) * 4;
      const int r = idx >> 6, c = idx & 63;     // r=key, c=dk
      const bf16* src = Vbase + (size_t)(kb + r) * DK + c;
      Vt[(c + 0) * 64 + r] = src[0];
      Vt[(c + 1) * 64 + r] = src[1];
      Vt[(c + 2) * 64 + r] = src[2];
      Vt[(c + 3) * 64 + r] = src[3];
    }
    // prefetch next V block while we compute on this one
    if (kb + 64 < S)
      __builtin_prefetch(Vbase + (size_t)(kb + 64) * DK + tid * 32, 0, 1);

    if (wave == 0) __builtin_amdgcn_s_wait_tensorcnt(0);
    __syncthreads();

    // ---- S = Q * K^T (16 x 64 per wave) ----
    floatx8 sc[4];
#pragma unroll
    for (int j = 0; j < 4; ++j) {
      floatx8 s = zero8();
      const bf16x16 b0 = load_frag_b(&Ks[(j * 16) * 64], 64);       // dk 0..31
      const bf16x16 b1 = load_frag_b(&Ks[(j * 16) * 64 + 32], 64);  // dk 32..63
      s = wmma_bf16(qa0, b0, s);
      s = wmma_bf16(qa1, b1, s);
      sc[j] = s;
    }

    // ---- online softmax (per query row; 16 lanes hold a row) ----
    constexpr float scale = 0.125f;             // 1/sqrt(64)
    float newm[8];
#pragma unroll
    for (int r = 0; r < 8; ++r) {
      float mx = -1e30f;
#pragma unroll
      for (int j = 0; j < 4; ++j) {
        sc[j][r] *= scale;
        mx = fmaxf(mx, sc[j][r]);
      }
#pragma unroll
      for (int off = 1; off < 16; off <<= 1)
        mx = fmaxf(mx, __shfl_xor(mx, off, 16));
      newm[r] = fmaxf(rm[r], mx);
    }
#pragma unroll
    for (int r = 0; r < 8; ++r) {
      const float corr = __expf(rm[r] - newm[r]);
      float ssum = 0.0f;
#pragma unroll
      for (int j = 0; j < 4; ++j) {
        const float p = __expf(sc[j][r] - newm[r]);
        sc[j][r] = p;
        ssum += p;
      }
#pragma unroll
      for (int off = 1; off < 16; off <<= 1)
        ssum += __shfl_xor(ssum, off, 16);
      rl[r] = rl[r] * corr + ssum;
      rm[r] = newm[r];
#pragma unroll
      for (int j = 0; j < 4; ++j) acc[j][r] *= corr;
    }

    // ---- stage P (C-layout) -> LDS -> A-layout fragments ----
    bf16* pw = Ps[wave];
#pragma unroll
    for (int j = 0; j < 4; ++j)
#pragma unroll
      for (int r = 0; r < 8; ++r)
        pw[(r + grp * 8) * PLD + j * 16 + nlo] = (bf16)sc[j][r];

    const bf16x16 pa0 = load_frag_a(pw, PLD);        // keys 0..31
    const bf16x16 pa1 = load_frag_a(pw + 32, PLD);   // keys 32..63

    // ---- O += P * V ----
#pragma unroll
    for (int j = 0; j < 4; ++j) {
      const bf16x16 vb0 = load_frag_b(&Vt[(j * 16) * 64], 64);       // keys 0..31
      const bf16x16 vb1 = load_frag_b(&Vt[(j * 16) * 64 + 32], 64);  // keys 32..63
      acc[j] = wmma_bf16(pa0, vb0, acc[j]);
      acc[j] = wmma_bf16(pa1, vb1, acc[j]);
    }
    __syncthreads();   // protects Ks/Vt before next TDM issue / reload
  }

  // ---- normalize and store O merged [B,S,1024] bf16 ----
  const int b = bh >> 4, h = bh & 15;
  const int srow = qb * 64 + wave * 16;
#pragma unroll
  for (int j = 0; j < 4; ++j)
#pragma unroll
    for (int r = 0; r < 8; ++r) {
      const int m = r + grp * 8;
      const float o = acc[j][r] / rl[r];
      const size_t off =
          ((size_t)b * 2048 + srow + m) * 1024 + h * 64 + j * 16 + nlo;
      O[off] = (bf16)o;
    }
}

// ---------------------------------------------------------------------------
extern "C" void kernel_launch(void* const* d_in, const int* in_sizes, int n_in,
                              void* d_out, int out_size, void* d_ws,
                              size_t ws_size, hipStream_t stream) {
  (void)in_sizes; (void)n_in; (void)out_size; (void)ws_size;
  constexpr int B = 4, S = 2048, D = 1024;
  constexpr int M = B * S;                      // 8192

  const float* q   = (const float*)d_in[0];
  const float* kin = (const float*)d_in[1];
  const float* v   = (const float*)d_in[2];
  const float* Wq  = (const float*)d_in[3];
  const float* bq  = (const float*)d_in[4];
  const float* Wk  = (const float*)d_in[5];
  const float* bk  = (const float*)d_in[6];
  const float* Wv  = (const float*)d_in[7];
  const float* bv  = (const float*)d_in[8];
  const float* Wo  = (const float*)d_in[9];
  const float* bo  = (const float*)d_in[10];

  const size_t tensorBytes = (size_t)M * D * sizeof(bf16);  // 16 MB
  char* ws = (char*)d_ws;
  bf16* Qw = (bf16*)(ws);
  bf16* Kw = (bf16*)(ws + tensorBytes);
  bf16* Vw = (bf16*)(ws + 2 * tensorBytes);
  bf16* Ow = (bf16*)(ws + 3 * tensorBytes);

  const dim3 gGrid(D / 128, M / 128);           // 8 x 64
  const dim3 gBlk(256);

  gemm128<false, true><<<gGrid, gBlk, 0, stream>>>(q,   Wq, bq, Qw, M, D, D);
  gemm128<false, true><<<gGrid, gBlk, 0, stream>>>(kin, Wk, bk, Kw, M, D, D);
  gemm128<false, true><<<gGrid, gBlk, 0, stream>>>(v,   Wv, bv, Vw, M, D, D);

  const dim3 aGrid(S / 64, B * 16);             // 32 x 64
  attn_kernel<<<aGrid, dim3(128), 0, stream>>>(Qw, Kw, Vw, Ow);

  gemm128<true, false><<<gGrid, gBlk, 0, stream>>>(Ow, Wo, bo, (float*)d_out,
                                                   M, D, D);
}
